// HistoryFilterGCN_3917010174745
// MI455X (gfx1250) — matrix operand order repaired and checked by reading, not compiled, not run on hardware
//
#include <hip/hip_runtime.h>
#include <hip/hip_bf16.h>
#include <math.h>

typedef __attribute__((ext_vector_type(16))) _Float16 v16h;
typedef __attribute__((ext_vector_type(8)))  float    v8f;
typedef __attribute__((ext_vector_type(8)))  int      v8i;

#define WMMA16(a, b, c) \
  __builtin_amdgcn_wmma_f32_16x16x32_f16(false, (a), false, (b), (short)0, (c), false, false)

__device__ __forceinline__ void lds_fence() {
  asm volatile("s_wait_dscnt 0" ::: "memory");
}

__device__ __forceinline__ v8f vzero8() {
  v8f z = {0.f, 0.f, 0.f, 0.f, 0.f, 0.f, 0.f, 0.f};
  return z;
}

// Branchless tanh: no EXEC-writing compares, ~7 VALU ops
// (v_exp_f32 + v_rcp_f32 + fma/min/max). Keeps EXEC all-ones around WMMA.
__device__ __forceinline__ float fast_tanh(float v) {
  float x = fminf(fmaxf(v, -15.f), 15.f);   // exp(2*15) ~ 1e13, no overflow
  float e = __expf(2.f * x);                // v_exp_f32
  return 1.f - 2.f * __builtin_amdgcn_rcpf(e + 1.f);
}

// ---- A operand: 16 x kdim row-major f16 tile in LDS (kdim even, rows 4B aligned).
// CDNA5 16-bit A layout: lane L -> M = L&15, half = L>>4;
// VGPR v holds K = (v>>2)*16 + half*8 + (v&3)*2 (+1).
__device__ __forceinline__ v16h load_A(const _Float16* smem, int kdim, int lane, int kbase) {
  const unsigned* p = (const unsigned*)smem;
  int m = lane & 15, half = lane >> 4;
  union { v16h h; v8i i; } ua;
#pragma unroll
  for (int v = 0; v < 8; ++v) {
    int k0 = kbase + ((v >> 2) << 4) + (half << 3) + ((v & 3) << 1);
    ua.i[v] = p[(m * kdim + k0) >> 1];
  }
  return ua.h;
}

// ---- B operand from LDS weights stored pair-interleaved:
// element (k, n) at f16 index (k>>1)*128 + 2*n + (k&1)  (== uint index (k>>1)*64 + n).
// CDNA5 B layout (column-major): lane L -> N = L&15, half = L>>4; VGPR v holds K = (v>>2)*16 + half*8 + (v&3)*2 (+1).
__device__ __forceinline__ v16h load_B(const _Float16* w, int lane, int kbase, int ntile) {
  const unsigned* p = (const unsigned*)w;
  int n = (lane & 15) + (ntile << 4);
  int tb = (kbase + ((lane >> 4) << 3)) >> 1;
  union { v16h h; v8i i; } ub;
#pragma unroll
  for (int v = 0; v < 8; ++v) {
    ub.i[v] = p[(tb + ((v >> 2) << 3) + (v & 3)) * 64 + n];
  }
  return ub.h;
}

// D = A(16 x kdim) * W(kdim x 64) accumulated into acc[4] (four 16-wide N tiles).
__device__ __forceinline__ void gemm4(const _Float16* A, int kdim, const _Float16* W,
                                      v8f acc[4], int lane) {
#pragma unroll
  for (int kb = 0; kb < 256; kb += 32) {       // bounded unroll; kdim <= 96 at call sites
    if (kb >= kdim) break;
    v16h a = load_A(A, kdim, lane, kb);
#pragma unroll
    for (int nt = 0; nt < 4; ++nt) {
      v16h b = load_B(W, lane, kb, nt);
      acc[nt] = WMMA16(a, b, acc[nt]);
    }
  }
}

// C layout: VGPR r -> M = r + 8*(lane>>4), N = (lane&15) + 16*nt.  tanh(acc+bias) -> 16x64 f16 act tile.
__device__ __forceinline__ void act_store(const v8f acc[4], const float* bias,
                                          _Float16* act, int lane) {
  int c0 = lane & 15, half = lane >> 4;
#pragma unroll
  for (int nt = 0; nt < 4; ++nt) {
    int n = c0 + (nt << 4);
    float bb = bias[n];
#pragma unroll
    for (int r = 0; r < 8; ++r) {
      act[(r + (half << 3)) * 64 + n] = (_Float16)fast_tanh(acc[nt][r] + bb);
    }
  }
}

// Stage f32 weight [krows][64] -> LDS f16 pair-interleaved, zero padded to kpad rows.
__device__ __forceinline__ void stage_weights(const float* w, int krows, int kpad,
                                              _Float16* sw, int tid, int nthreads) {
  for (int i = tid; i < kpad * 64; i += nthreads) {
    int k = i >> 6, n = i & 63;
    float v = (k < krows) ? w[k * 64 + n] : 0.f;
    sw[((k >> 1) * 64 + n) * 2 + (k & 1)] = (_Float16)v;
  }
}

// ---------------------------------------------------------------- zero workspace
__global__ void k_zero(float* __restrict__ p, long long n) {
  long long i = (long long)blockIdx.x * blockDim.x + threadIdx.x;
  long long st = (long long)gridDim.x * blockDim.x;
  for (; i < n; i += st) p[i] = 0.f;
}

// ---------------------------------------------------------------- a2s edge MLP + segment sum
__global__ __launch_bounds__(128) void k_a2s(
    const float* __restrict__ pos_state, const float* __restrict__ pos_action,
    const float* __restrict__ u, const int* __restrict__ src, const int* __restrict__ dst,
    const float* __restrict__ dis,
    const float* __restrict__ w1, const float* __restrict__ b1,
    const float* __restrict__ w2, const float* __restrict__ b2,
    const float* __restrict__ w3, const float* __restrict__ b3,
    float* __restrict__ sum_u, int E) {
  __shared__ __align__(16) _Float16 sW1[32 * 64];
  __shared__ __align__(16) _Float16 sW2[64 * 64];
  __shared__ __align__(16) _Float16 sW3[64 * 64];
  __shared__ float sB1[64], sB2[64], sB3[64];
  __shared__ __align__(16) _Float16 sIn[4][16 * 32];
  __shared__ __align__(16) _Float16 sAct[4][16 * 64];

  int t = threadIdx.x;
  stage_weights(w1, 13, 32, sW1, t, 128);
  stage_weights(w2, 64, 64, sW2, t, 128);
  stage_weights(w3, 64, 64, sW3, t, 128);
  if (t < 64) { sB1[t] = b1[t]; sB2[t] = b2[t]; sB3[t] = b3[t]; }
  __syncthreads();

  int lane = t & 31, wid = t >> 5;
  int m = lane & 15, half = lane >> 4;
  _Float16* inT = sIn[wid];
  _Float16* act = sAct[wid];
  int gw = blockIdx.x * 4 + wid;
  int nw = gridDim.x * 4;
  int ntiles = (E + 15) >> 4;

  for (int tile = gw; tile < ntiles; tile += nw) {
    int base = tile << 4;
    int row = base + m; if (row >= E) row = E - 1;
    int s = src[row], d = dst[row];
    // features: [pos_action[s](2), pos_state[d](2), dis(1), u[s](8)] pad->32
#pragma unroll
    for (int j = 0; j < 16; ++j) {
      int k = (half << 4) + j;
      float v = 0.f;
      if (k < 2)       v = pos_action[s * 2 + k];
      else if (k < 4)  v = pos_state[d * 2 + (k - 2)];
      else if (k < 5)  v = dis[row];
      else if (k < 13) v = u[s * 8 + (k - 5)];
      inT[m * 32 + k] = (_Float16)v;
    }
    lds_fence();

    v8f a1[4]; for (int i = 0; i < 4; ++i) a1[i] = vzero8();
    gemm4(inT, 32, sW1, a1, lane);
    act_store(a1, sB1, act, lane);
    lds_fence();

    v8f a2[4]; for (int i = 0; i < 4; ++i) a2[i] = vzero8();
    gemm4(act, 64, sW2, a2, lane);
    act_store(a2, sB2, act, lane);
    lds_fence();

    v8f a3[4]; for (int i = 0; i < 4; ++i) a3[i] = vzero8();
    gemm4(act, 64, sW3, a3, lane);

#pragma unroll
    for (int r = 0; r < 8; ++r) {
      int grow = base + r + (half << 3);
      if (grow < E) {
        int dd = dst[grow];
#pragma unroll
        for (int nt = 0; nt < 4; ++nt) {
          int n = m + (nt << 4);
          atomicAdd(&sum_u[dd * 64 + n], a3[nt][r] + sB3[n]);
        }
      }
    }
  }
}

// ---------------------------------------------------------------- s2s edge MLP + segment sum + degree
__global__ __launch_bounds__(128) void k_s2s(
    const float* __restrict__ pos_state, const float* __restrict__ x,
    const float* __restrict__ h, const int* __restrict__ src, const int* __restrict__ dst,
    const float* __restrict__ dis,
    const float* __restrict__ w1, const float* __restrict__ b1,
    const float* __restrict__ w2, const float* __restrict__ b2,
    const float* __restrict__ w3, const float* __restrict__ b3,
    float* __restrict__ sum_x, float* __restrict__ deg, int E) {
  __shared__ __align__(16) _Float16 sW1[96 * 64];
  __shared__ __align__(16) _Float16 sW2[64 * 64];
  __shared__ __align__(16) _Float16 sW3[64 * 64];
  __shared__ float sB1[64], sB2[64], sB3[64];
  __shared__ __align__(16) _Float16 sIn[4][16 * 96];
  __shared__ __align__(16) _Float16 sAct[4][16 * 64];

  int t = threadIdx.x;
  stage_weights(w1, 77, 96, sW1, t, 128);
  stage_weights(w2, 64, 64, sW2, t, 128);
  stage_weights(w3, 64, 64, sW3, t, 128);
  if (t < 64) { sB1[t] = b1[t]; sB2[t] = b2[t]; sB3[t] = b3[t]; }
  __syncthreads();

  int lane = t & 31, wid = t >> 5;
  int m = lane & 15, half = lane >> 4;
  _Float16* inT = sIn[wid];
  _Float16* act = sAct[wid];
  int gw = blockIdx.x * 4 + wid;
  int nw = gridDim.x * 4;
  int ntiles = (E + 15) >> 4;

  for (int tile = gw; tile < ntiles; tile += nw) {
    int base = tile << 4;
    if (lane < 16 && base + lane < E) atomicAdd(&deg[dst[base + lane]], 1.0f);

    int row = base + m; if (row >= E) row = E - 1;
    int s = src[row], d = dst[row];
    // features: [pos_state[s](2), pos_state[d](2), dis(1), x[s](8), h[s](64)] pad->96
#pragma unroll
    for (int c = 0; c < 3; ++c) {
#pragma unroll
      for (int j = 0; j < 16; ++j) {
        int k = c * 32 + (half << 4) + j;
        float v = 0.f;
        if (k < 2)       v = pos_state[s * 2 + k];
        else if (k < 4)  v = pos_state[d * 2 + (k - 2)];
        else if (k < 5)  v = dis[row];
        else if (k < 13) v = x[s * 8 + (k - 5)];
        else if (k < 77) v = h[s * 64 + (k - 13)];
        inT[m * 96 + k] = (_Float16)v;
      }
    }
    lds_fence();

    v8f a1[4]; for (int i = 0; i < 4; ++i) a1[i] = vzero8();
    gemm4(inT, 96, sW1, a1, lane);
    act_store(a1, sB1, act, lane);
    lds_fence();

    v8f a2[4]; for (int i = 0; i < 4; ++i) a2[i] = vzero8();
    gemm4(act, 64, sW2, a2, lane);
    act_store(a2, sB2, act, lane);
    lds_fence();

    v8f a3[4]; for (int i = 0; i < 4; ++i) a3[i] = vzero8();
    gemm4(act, 64, sW3, a3, lane);

#pragma unroll
    for (int r = 0; r < 8; ++r) {
      int grow = base + r + (half << 3);
      if (grow < E) {
        int dd = dst[grow];
#pragma unroll
        for (int nt = 0; nt < 4; ++nt) {
          int n = m + (nt << 4);
          atomicAdd(&sum_x[dd * 64 + n], a3[nt][r] + sB3[n]);
        }
      }
    }
  }
}

// ---------------------------------------------------------------- node update MLP
__global__ __launch_bounds__(128) void k_upd(
    const float* __restrict__ pos_state, const float* __restrict__ h,
    const float* __restrict__ x, const float* __restrict__ sum_u,
    const float* __restrict__ sum_x, const float* __restrict__ deg,
    const float* __restrict__ w1, const float* __restrict__ b1,
    const float* __restrict__ w2, const float* __restrict__ b2,
    const float* __restrict__ w3, const float* __restrict__ b3,
    float* __restrict__ out, int NS) {
  __shared__ __align__(16) _Float16 sW1[224 * 64];
  __shared__ __align__(16) _Float16 sW2[64 * 64];
  __shared__ __align__(16) _Float16 sW3[64 * 64];
  __shared__ float sB1[64], sB2[64], sB3[64];
  __shared__ __align__(16) _Float16 sIn[4][16 * 32];   // streamed K-chunk
  __shared__ __align__(16) _Float16 sAct[4][16 * 64];

  int t = threadIdx.x;
  stage_weights(w1, 202, 224, sW1, t, 128);
  stage_weights(w2, 64, 64, sW2, t, 128);
  stage_weights(w3, 64, 64, sW3, t, 128);
  if (t < 64) { sB1[t] = b1[t]; sB2[t] = b2[t]; sB3[t] = b3[t]; }
  __syncthreads();

  int lane = t & 31, wid = t >> 5;
  int m = lane & 15, half = lane >> 4;
  _Float16* inT = sIn[wid];
  _Float16* act = sAct[wid];
  int gw = blockIdx.x * 4 + wid;
  int nw = gridDim.x * 4;
  int ntiles = (NS + 15) >> 4;

  for (int tile = gw; tile < ntiles; tile += nw) {
    int base = tile << 4;
    int node = base + m; if (node >= NS) node = NS - 1;
    float inv = __builtin_amdgcn_rcpf(fmaxf(deg[node], 1.0f));

    // layer 1: K = 202 padded to 224, streamed in 7 chunks of 32
    v8f a1[4]; for (int i = 0; i < 4; ++i) a1[i] = vzero8();
    for (int kc = 0; kc < 7; ++kc) {
      int kb = kc * 32;
#pragma unroll
      for (int j = 0; j < 16; ++j) {
        int kk = (half << 4) + j;
        int k = kb + kk;
        float v = 0.f;
        if (k < 2)        v = pos_state[node * 2 + k];
        else if (k < 66)  v = h[node * 64 + (k - 2)];
        else if (k < 130) v = sum_u[node * 64 + (k - 66)];
        else if (k < 194) v = sum_x[node * 64 + (k - 130)] * inv;
        else if (k < 202) v = x[node * 8 + (k - 194)];
        inT[m * 32 + kk] = (_Float16)v;
      }
      lds_fence();
      v16h a = load_A(inT, 32, lane, 0);
#pragma unroll
      for (int nt = 0; nt < 4; ++nt) {
        v16h b = load_B(sW1, lane, kb, nt);
        a1[nt] = WMMA16(a, b, a1[nt]);
      }
      lds_fence();
    }
    act_store(a1, sB1, act, lane);
    lds_fence();

    v8f a2[4]; for (int i = 0; i < 4; ++i) a2[i] = vzero8();
    gemm4(act, 64, sW2, a2, lane);
    act_store(a2, sB2, act, lane);
    lds_fence();

    v8f a3[4]; for (int i = 0; i < 4; ++i) a3[i] = vzero8();
    gemm4(act, 64, sW3, a3, lane);

#pragma unroll
    for (int r = 0; r < 8; ++r) {
      int gnode = base + r + (half << 3);
      if (gnode < NS) {
#pragma unroll
        for (int nt = 0; nt < 4; ++nt) {
          int n = m + (nt << 4);
          out[gnode * 64 + n] = a3[nt][r] + sB3[n];
        }
      }
    }
  }
}

// ---------------------------------------------------------------- launcher
extern "C" void kernel_launch(void* const* d_in, const int* in_sizes, int n_in,
                              void* d_out, int out_size, void* d_ws, size_t ws_size,
                              hipStream_t stream) {
  const float* pos_state  = (const float*)d_in[0];
  const float* pos_action = (const float*)d_in[1];
  const float* h          = (const float*)d_in[2];
  const float* x          = (const float*)d_in[3];
  const float* u          = (const float*)d_in[4];
  const int*   a2s_src    = (const int*)d_in[5];
  const int*   a2s_dst    = (const int*)d_in[6];
  const float* a2s_dis    = (const float*)d_in[7];
  const int*   s2s_src    = (const int*)d_in[8];
  const int*   s2s_dst    = (const int*)d_in[9];
  const float* s2s_dis    = (const float*)d_in[10];
  const float* u2h_w1 = (const float*)d_in[11]; const float* u2h_b1 = (const float*)d_in[12];
  const float* u2h_w2 = (const float*)d_in[13]; const float* u2h_b2 = (const float*)d_in[14];
  const float* u2h_w3 = (const float*)d_in[15]; const float* u2h_b3 = (const float*)d_in[16];
  const float* x2h_w1 = (const float*)d_in[17]; const float* x2h_b1 = (const float*)d_in[18];
  const float* x2h_w2 = (const float*)d_in[19]; const float* x2h_b2 = (const float*)d_in[20];
  const float* x2h_w3 = (const float*)d_in[21]; const float* x2h_b3 = (const float*)d_in[22];
  const float* upd_w1 = (const float*)d_in[23]; const float* upd_b1 = (const float*)d_in[24];
  const float* upd_w2 = (const float*)d_in[25]; const float* upd_b2 = (const float*)d_in[26];
  const float* upd_w3 = (const float*)d_in[27]; const float* upd_b3 = (const float*)d_in[28];

  int NSv = in_sizes[0] / 2;
  int EAv = in_sizes[5];
  int ESv = in_sizes[8];

  float* sum_u = (float*)d_ws;
  float* sum_x = sum_u + (size_t)NSv * 64;
  float* degv  = sum_x + (size_t)NSv * 64;

  long long zn = (long long)NSv * 128 + NSv;
  k_zero<<<512, 256, 0, stream>>>(sum_u, zn);

  int ta = (EAv + 15) / 16, ba = (ta + 3) / 4;
  k_a2s<<<ba, 128, 0, stream>>>(pos_state, pos_action, u, a2s_src, a2s_dst, a2s_dis,
                                u2h_w1, u2h_b1, u2h_w2, u2h_b2, u2h_w3, u2h_b3,
                                sum_u, EAv);

  int ts = (ESv + 15) / 16, bs = (ts + 3) / 4;
  k_s2s<<<bs, 128, 0, stream>>>(pos_state, x, h, s2s_src, s2s_dst, s2s_dis,
                                x2h_w1, x2h_b1, x2h_w2, x2h_b2, x2h_w3, x2h_b3,
                                sum_x, degv, ESv);

  int tn = (NSv + 15) / 16, bn = (tn + 3) / 4;
  k_upd<<<bn, 128, 0, stream>>>(pos_state, h, x, sum_u, sum_x, degv,
                                upd_w1, upd_b1, upd_w2, upd_b2, upd_w3, upd_b3,
                                (float*)d_out, NSv);
}
